// HeteroGNN_Edge_59923383714578
// MI455X (gfx1250) — compile-verified, matrix operand order checked by reading, not run on hardware
//
#include <hip/hip_runtime.h>
#include <hip/hip_bf16.h>

// ---------------------------------------------------------------------------
// HeteroGNN (GNN4ID-style) for MI455X / gfx1250, wave32, WMMA f16 matrix cores
// ---------------------------------------------------------------------------

typedef __attribute__((ext_vector_type(16))) _Float16 v16h;
typedef __attribute__((ext_vector_type(8)))  float    v8f;

#define H_DIM    64
#define NFLOW    8192
#define NPACKET  262144
#define NGRAPH   8192
#define E_FP     262144
#define E_PF     262144
#define E_PP     1048576
#define BN_EPS   1e-5f
#define ENC_NEG_INF 0x007FFFFFu   // fenc(-inf)

// monotonic float <-> uint mapping so atomicMax(uint) == max(float)
static __device__ __forceinline__ unsigned fenc(float f) {
  unsigned u = __float_as_uint(f);
  return (u & 0x80000000u) ? ~u : (u | 0x80000000u);
}
static __device__ __forceinline__ float fdec(unsigned u) {
  unsigned b = (u & 0x80000000u) ? (u & 0x7fffffffu) : ~u;
  return __uint_as_float(b);
}

// ------------------------------- fill --------------------------------------
__global__ void fill_u32(unsigned* __restrict__ p, unsigned v, long n) {
  long i = (long)blockIdx.x * blockDim.x + threadIdx.x;
  if (i < n) p[i] = v;
}

// ------------------------- W prepack (f16 fragments) -----------------------
// Wp layout: [tile(0..3)][chunk(0..K/32-1)][lane(0..31)][elem(0..15)] f16,
// matching the 16-bit B-matrix 32x16 WMMA operand layout.
__global__ void pack_w(const float* __restrict__ W, _Float16* __restrict__ Wp, int K) {
  int nchunk = K >> 5;
  int total  = 4 * nchunk * 512;
  int t = blockIdx.x * blockDim.x + threadIdx.x;
  if (t >= total) return;
  int e     = t & 15;
  int lane  = (t >> 4) & 31;
  int rest  = t >> 9;
  int chunk = rest % nchunk;
  int tile  = rest / nchunk;
  int klocal = e + (e >= 8 ? 8 : 0) + (lane >= 16 ? 8 : 0);
  int k = chunk * 32 + klocal;
  int n = tile * 16 + (lane & 15);
  Wp[t] = (_Float16)W[k * H_DIM + n];
}

// ------------------------- GEMM: C[M x 64] = A[M x K] @ W --------------------
// 128 threads = 4 waves/block; each wave owns a 16x64 output tile.
// Packed W fragments staged once per block into LDS (ds_load_b128 per use);
// A stream prefetched one K-chunk ahead (global_prefetch).
// K in {32,64,128}; M multiple of 64. EXEC stays all-ones around WMMA.
__global__ void gemm_n64(const float* __restrict__ A, const _Float16* __restrict__ Wp,
                         float* __restrict__ C, int M, int K) {
  __shared__ _Float16 sW[8192];                    // up to 128x64 fragments (16 KB)
  int tid  = threadIdx.x;
  int lane = tid & 31;
  int wv   = tid >> 5;
  int nchunk = K >> 5;

  // cooperative staging: nchunk*4 KB, 16B vectors
  {
    const float4* srcv = (const float4*)Wp;
    float4* dstv = (float4*)sW;
    int n4 = nchunk * 256;
    for (int i = tid; i < n4; i += 128) dstv[i] = srcv[i];
  }
  __syncthreads();

  long row0 = (long)blockIdx.x * 64 + wv * 16;
  long row  = row0 + (lane & 15);
  int  kh   = (lane >> 4) << 3;                    // lanes 16..31 take K offset +8
  const float* arow = A + row * K;
  int tstride = nchunk * 512;

  v8f c0 = {}, c1 = {}, c2 = {}, c3 = {};
  for (int ch = 0; ch < nchunk; ++ch) {
    const float* ap = arow + ch * 32 + kh;
    if (ch + 1 < nchunk) __builtin_prefetch(ap + 32, 0, 3);   // global_prefetch A
    v16h af;
#pragma unroll
    for (int e = 0; e < 8; ++e) af[e]     = (_Float16)ap[e];        // K: kh+0..7
#pragma unroll
    for (int e = 0; e < 8; ++e) af[e + 8] = (_Float16)ap[16 + e];   // K: kh+16..23
    const _Float16* bb = sW + (ch * 32 + lane) * 16;
    v16h b0 = *(const v16h*)(bb);
    v16h b1 = *(const v16h*)(bb + tstride);
    v16h b2 = *(const v16h*)(bb + 2 * tstride);
    v16h b3 = *(const v16h*)(bb + 3 * tstride);
    c0 = __builtin_amdgcn_wmma_f32_16x16x32_f16(false, af, false, b0, (short)0, c0, false, false);
    c1 = __builtin_amdgcn_wmma_f32_16x16x32_f16(false, af, false, b1, (short)0, c1, false, false);
    c2 = __builtin_amdgcn_wmma_f32_16x16x32_f16(false, af, false, b2, (short)0, c2, false, false);
    c3 = __builtin_amdgcn_wmma_f32_16x16x32_f16(false, af, false, b3, (short)0, c3, false, false);
  }
  int rofs = (lane >> 4) << 3;
  int ncol = lane & 15;
#pragma unroll
  for (int r = 0; r < 8; ++r) {
    float* cp = C + (row0 + r + rofs) * H_DIM + ncol;
    cp[0]  = c0[r];
    cp[16] = c1[r];
    cp[32] = c2[r];
    cp[48] = c3[r];
  }
}

// ------------------------- rowdot: out[i] = X[i,:] . v ----------------------
__global__ void rowdot(const float* __restrict__ X, const float* __restrict__ v,
                       float* __restrict__ out, int M, int K) {
  int gw   = (int)((blockIdx.x * (long)blockDim.x + threadIdx.x) >> 5);
  int lane = threadIdx.x & 31;
  if (gw >= M) return;
  const float* row = X + (long)gw * K;
  float acc = 0.f;
  for (int k = lane; k < K; k += 32) acc += row[k] * v[k];
#pragma unroll
  for (int off = 16; off > 0; off >>= 1) acc += __shfl_down(acc, off, 32);
  if (lane == 0) out[gw] = acc;
}

// ------------------------- edge phase ---------------------------------------
__global__ void edge_logit_max(const int* __restrict__ src, const int* __restrict__ dst,
                               const float* __restrict__ as_, const float* __restrict__ ad_,
                               const float* __restrict__ ae_, float* __restrict__ abuf,
                               unsigned* __restrict__ amax, int E) {
  int e = blockIdx.x * blockDim.x + threadIdx.x;
  if (e >= E) return;
  float a = as_[src[e]] + ad_[dst[e]] + ae_[e];
  a = (a >= 0.f) ? a : 0.2f * a;            // LeakyReLU(0.2) on logits
  abuf[e] = a;
  atomicMax(amax + dst[e], fenc(a));
}

__global__ void edge_exp_sum(const int* __restrict__ dst, const float* __restrict__ abuf,
                             const unsigned* __restrict__ amax, float* __restrict__ den,
                             float* __restrict__ exb, int E) {
  int e = blockIdx.x * blockDim.x + threadIdx.x;
  if (e >= E) return;
  int d = dst[e];
  float ex = __expf(abuf[e] - fdec(amax[d]));
  exb[e] = ex;
  atomicAdd(den + d, ex);
}

// one wave per edge; each lane scatters 2 of the 64 features
__global__ void edge_scatter(const int* __restrict__ src, const int* __restrict__ dst,
                             const float* __restrict__ exb, const float* __restrict__ den,
                             const float* __restrict__ hs, float* __restrict__ out, int E) {
  int gw   = (int)((blockIdx.x * (long)blockDim.x + threadIdx.x) >> 5);
  int lane = threadIdx.x & 31;
  if (gw >= E) return;
  int d = dst[gw], s = src[gw];
  float w = exb[gw] / fmaxf(den[d], 1e-16f);
  const float2 hv = *(const float2*)(hs + (long)s * H_DIM + lane * 2);
  float* op = out + (long)d * H_DIM + lane * 2;
  atomicAdd(op,     w * hv.x);
  atomicAdd(op + 1, w * hv.y);
}

// ------------------------- batchnorm ----------------------------------------
__global__ void bn_stats(const float* __restrict__ X, long M, float* __restrict__ sums) {
  int c = threadIdx.x & 63;
  long stride = (long)gridDim.x * blockDim.x;          // multiple of 64
  long total  = M * H_DIM;
  float s = 0.f, s2 = 0.f;
  for (long i = (long)blockIdx.x * blockDim.x + threadIdx.x; i < total; i += stride) {
    float x = X[i]; s += x; s2 += x * x;
  }
  atomicAdd(sums + c, s);
  atomicAdd(sums + 64 + c, s2);
}

__global__ void bn_apply(float* __restrict__ X, const float* __restrict__ sums,
                         const float* __restrict__ g, const float* __restrict__ b, long M) {
  long i = (long)blockIdx.x * blockDim.x + threadIdx.x;
  if (i >= M * H_DIM) return;
  int c = (int)(i & 63);
  float mu  = sums[c] / (float)M;
  float var = sums[64 + c] / (float)M - mu * mu;
  float y = g[c] * (X[i] - mu) * rsqrtf(var + BN_EPS) + b[c];
  X[i] = (y >= 0.f) ? y : 0.01f * y;                   // LeakyReLU(0.01)
}

// ------------------------- pooling ------------------------------------------
__global__ void pool_scatter(const float* __restrict__ X, const int* __restrict__ batch,
                             float* __restrict__ pool, float* __restrict__ cnt, int M) {
  int gw   = (int)((blockIdx.x * (long)blockDim.x + threadIdx.x) >> 5);
  int lane = threadIdx.x & 31;
  if (gw >= M) return;
  int g = batch[gw];
  const float2 hv = *(const float2*)(X + (long)gw * H_DIM + lane * 2);
  float* pp = pool + (long)g * H_DIM + lane * 2;
  atomicAdd(pp,     hv.x);
  atomicAdd(pp + 1, hv.y);
  if (lane == 0) atomicAdd(cnt + g, 1.0f);
}

__global__ void emb_build(const float* __restrict__ poolf, const float* __restrict__ cntf,
                          const float* __restrict__ poolp, const float* __restrict__ cntp,
                          float* __restrict__ emb) {
  int t = blockIdx.x * blockDim.x + threadIdx.x;   // NGRAPH*128 threads
  int g = t >> 7, c = t & 127;
  float v = (c < 64) ? poolf[(long)g * 64 + c] / fmaxf(cntf[g], 1.f)
                     : poolp[(long)g * 64 + (c - 64)] / fmaxf(cntp[g], 1.f);
  emb[t] = v;
}

// ------------------------- MLP collapse + final -----------------------------
__global__ void mlp_t1(const float* __restrict__ W1, const float* __restrict__ W2,
                       float* __restrict__ T1) {       // T1[128x16] = W1@W2
  int t = blockIdx.x * blockDim.x + threadIdx.x;
  if (t >= 128 * 16) return;
  int i = t / 16, k = t % 16;
  float a = 0.f;
  for (int j = 0; j < 64; ++j) a += W1[i * 64 + j] * W2[j * 16 + k];
  T1[t] = a;
}
__global__ void mlp_weff(const float* __restrict__ T1, const float* __restrict__ W3,
                         float* __restrict__ Weff) {   // Weff[128x8] = T1@W3
  int t = blockIdx.x * blockDim.x + threadIdx.x;
  if (t >= 128 * 8) return;
  int i = t / 8, k = t % 8;
  float a = 0.f;
  for (int j = 0; j < 16; ++j) a += T1[i * 16 + j] * W3[j * 8 + k];
  Weff[t] = a;
}
__global__ void mlp_beff(const float* __restrict__ b1, const float* __restrict__ W2,
                         const float* __restrict__ b2, const float* __restrict__ W3,
                         const float* __restrict__ b3, float* __restrict__ beff) {
  int k = threadIdx.x;
  if (k >= 8) return;
  float acc = b3[k];
  for (int j = 0; j < 16; ++j) {
    float tj = b2[j];
    for (int i = 0; i < 64; ++i) tj += b1[i] * W2[i * 16 + j];
    acc += tj * W3[j * 8 + k];
  }
  beff[k] = acc;
}
__global__ void mlp_final(const float* __restrict__ emb, const float* __restrict__ Weff,
                          const float* __restrict__ beff, float* __restrict__ out) {
  int g = blockIdx.x * blockDim.x + threadIdx.x;
  if (g >= NGRAPH) return;
  float acc[8];
#pragma unroll
  for (int k = 0; k < 8; ++k) acc[k] = beff[k];
  const float* er = emb + (long)g * 128;
  for (int j = 0; j < 128; ++j) {
    float x = er[j];
    const float* wr = Weff + j * 8;
#pragma unroll
    for (int k = 0; k < 8; ++k) acc[k] += x * wr[k];
  }
  float mx = acc[0];
#pragma unroll
  for (int k = 1; k < 8; ++k) mx = fmaxf(mx, acc[k]);
  float s = 0.f;
#pragma unroll
  for (int k = 0; k < 8; ++k) s += __expf(acc[k] - mx);
  float l = mx + __logf(s);
#pragma unroll
  for (int k = 0; k < 8; ++k) out[(long)g * 8 + k] = acc[k] - l;
}

// ---------------------------------------------------------------------------
// host-side orchestration
// ---------------------------------------------------------------------------
static inline void fillN(hipStream_t st, void* p, unsigned v, long n) {
  fill_u32<<<(unsigned)((n + 255) / 256), 256, 0, st>>>((unsigned*)p, v, n);
}

static void run_gat(hipStream_t st,
                    const float* xsrc, int Ns, int dsrc,
                    const float* xdst, int Nd, int ddst,
                    const int* ei, int E, const float* ea,
                    const float* Wsrc, const float* Wdst, const float* Wedge,
                    const float* att_src, const float* att_dst, const float* att_edge,
                    float* HS, float* ASB, float* ADB, float* AEB,
                    float* ABUF, float* EXB, unsigned* AMAX, float* DEN,
                    _Float16* WP, float* UD, float* UE,
                    float* out /* accumulated */) {
  const int* src = ei;
  const int* dst = ei + E;
  int nchunk = dsrc >> 5;
  int wtot = 4 * nchunk * 512;
  pack_w<<<(wtot + 255) / 256, 256, 0, st>>>(Wsrc, WP, dsrc);
  gemm_n64<<<Ns / 64, 128, 0, st>>>(xsrc, WP, HS, Ns, dsrc);                 // hs
  rowdot<<<(Ns + 7) / 8, 256, 0, st>>>(HS, att_src, ASB, Ns, H_DIM);         // as
  rowdot<<<(ddst + 7) / 8, 256, 0, st>>>(Wdst, att_dst, UD, ddst, H_DIM);    // u_dst
  rowdot<<<(Nd + 7) / 8, 256, 0, st>>>(xdst, UD, ADB, Nd, ddst);             // ad
  rowdot<<<(32 + 7) / 8, 256, 0, st>>>(Wedge, att_edge, UE, 32, H_DIM);      // u_edge
  rowdot<<<(E + 7) / 8, 256, 0, st>>>(ea, UE, AEB, E, 32);                   // ae
  fillN(st, AMAX, ENC_NEG_INF, Nd);
  fillN(st, DEN, 0u, Nd);
  edge_logit_max<<<(E + 255) / 256, 256, 0, st>>>(src, dst, ASB, ADB, AEB, ABUF, AMAX, E);
  edge_exp_sum<<<(E + 255) / 256, 256, 0, st>>>(dst, ABUF, AMAX, DEN, EXB, E);
  edge_scatter<<<E / 8, 256, 0, st>>>(src, dst, EXB, DEN, HS, out, E);
  // note: GATConv bias is omitted — it cancels exactly under the following BatchNorm
}

static void run_bn(hipStream_t st, float* X, long M,
                   const float* g, const float* b, float* SUMS) {
  fillN(st, SUMS, 0u, 128);
  bn_stats<<<1024, 256, 0, st>>>(X, M, SUMS);
  bn_apply<<<(unsigned)((M * H_DIM + 255) / 256), 256, 0, st>>>(X, SUMS, g, b, M);
}

extern "C" void kernel_launch(void* const* d_in, const int* in_sizes, int n_in,
                              void* d_out, int out_size, void* d_ws, size_t ws_size,
                              hipStream_t stream) {
  (void)in_sizes; (void)n_in; (void)out_size; (void)ws_size;
  // ---- data inputs (setup_inputs dict order) ----
  const float* x_flow       = (const float*)d_in[0];
  const float* x_packet     = (const float*)d_in[1];
  const int*   ei_fp        = (const int*)d_in[2];
  const int*   ei_pf        = (const int*)d_in[3];
  const int*   ei_pp        = (const int*)d_in[4];
  const float* ea_fp        = (const float*)d_in[5];
  const float* ea_pf        = (const float*)d_in[6];
  const float* ea_pp        = (const float*)d_in[7];
  const int*   batch_flow   = (const int*)d_in[8];
  const int*   batch_packet = (const int*)d_in[9];
  // ---- params: jax tree-flatten order (dict keys sorted alphabetically) ----
  // bn1: flow(g,b), packet(g,b); bn2 same; l1/l2: fp,pf,pp each
  // [W_dst, W_edge, W_src, att_dst, att_edge, att_src, bias]; mlp: W1,W2,W3,b1,b2,b3
  const float* bn1_fg = (const float*)d_in[10];
  const float* bn1_fb = (const float*)d_in[11];
  const float* bn1_pg = (const float*)d_in[12];
  const float* bn1_pb = (const float*)d_in[13];
  const float* bn2_fg = (const float*)d_in[14];
  const float* bn2_fb = (const float*)d_in[15];
  const float* bn2_pg = (const float*)d_in[16];
  const float* bn2_pb = (const float*)d_in[17];
  const int l1fp = 18, l1pf = 25, l1pp = 32, l2fp = 39, l2pf = 46, l2pp = 53;
#define GP(base, k) ((const float*)d_in[(base) + (k)])   // k: 0=W_dst 1=W_edge 2=W_src 3=att_dst 4=att_edge 5=att_src 6=bias
  const float* mW1 = (const float*)d_in[60];
  const float* mW2 = (const float*)d_in[61];
  const float* mW3 = (const float*)d_in[62];
  const float* mb1 = (const float*)d_in[63];
  const float* mb2 = (const float*)d_in[64];
  const float* mb3 = (const float*)d_in[65];

  // ---- workspace carve (floats) ----
  float* w = (float*)d_ws;
  size_t off = 0;
  auto alloc = [&](size_t n) { float* p = w + off; off += n; return p; };
  float*    HS    = alloc((size_t)NPACKET * H_DIM);
  float*    OUTP  = alloc((size_t)NPACKET * H_DIM);   // h_p (layer1), in-place BN
  float*    OUTP2 = alloc((size_t)NPACKET * H_DIM);   // h_p2
  float*    OUTF  = alloc((size_t)NFLOW * H_DIM);     // h_f
  float*    OUTF2 = alloc((size_t)NFLOW * H_DIM);     // h_f2
  float*    ABUF  = alloc(E_PP);
  float*    EXB   = alloc(E_PP);
  float*    AEB   = alloc(E_PP);
  float*    ASB   = alloc(NPACKET);
  float*    ADB   = alloc(NPACKET);
  unsigned* AMAX  = (unsigned*)alloc(NPACKET);
  float*    DEN   = alloc(NPACKET);
  float*    POOLF = alloc((size_t)NGRAPH * H_DIM);
  float*    CNTF  = alloc(NGRAPH);
  float*    POOLP = alloc((size_t)NGRAPH * H_DIM);
  float*    CNTP  = alloc(NGRAPH);
  float*    EMB   = alloc((size_t)NGRAPH * 128);
  float*    SUMS  = alloc(128);
  float*    UD    = alloc(128);
  float*    UE    = alloc(32);
  float*    T1    = alloc(2048);
  float*    WEFF  = alloc(1024);
  float*    BEFF  = alloc(8);
  _Float16* WP    = (_Float16*)alloc(4096);           // 128x64 f16 fragments

  // ---- zero accumulators ----
  fillN(stream, OUTP,  0u, (long)NPACKET * H_DIM);
  fillN(stream, OUTP2, 0u, (long)NPACKET * H_DIM);
  fillN(stream, OUTF,  0u, (long)NFLOW * H_DIM);
  fillN(stream, OUTF2, 0u, (long)NFLOW * H_DIM);

  // ---- HeteroConv layer 1 ----
  run_gat(stream, x_flow, NFLOW, 128, x_packet, NPACKET, 64, ei_fp, E_FP, ea_fp,
          GP(l1fp,2), GP(l1fp,0), GP(l1fp,1), GP(l1fp,5), GP(l1fp,3), GP(l1fp,4),
          HS, ASB, ADB, AEB, ABUF, EXB, AMAX, DEN, WP, UD, UE, OUTP);
  run_gat(stream, x_packet, NPACKET, 64, x_packet, NPACKET, 64, ei_pp, E_PP, ea_pp,
          GP(l1pp,2), GP(l1pp,0), GP(l1pp,1), GP(l1pp,5), GP(l1pp,3), GP(l1pp,4),
          HS, ASB, ADB, AEB, ABUF, EXB, AMAX, DEN, WP, UD, UE, OUTP);
  run_gat(stream, x_packet, NPACKET, 64, x_flow, NFLOW, 128, ei_pf, E_PF, ea_pf,
          GP(l1pf,2), GP(l1pf,0), GP(l1pf,1), GP(l1pf,5), GP(l1pf,3), GP(l1pf,4),
          HS, ASB, ADB, AEB, ABUF, EXB, AMAX, DEN, WP, UD, UE, OUTF);
  run_bn(stream, OUTP, NPACKET, bn1_pg, bn1_pb, SUMS);
  run_bn(stream, OUTF, NFLOW,   bn1_fg, bn1_fb, SUMS);

  // ---- HeteroConv layer 2 ----
  run_gat(stream, OUTF, NFLOW, 64, OUTP, NPACKET, 64, ei_fp, E_FP, ea_fp,
          GP(l2fp,2), GP(l2fp,0), GP(l2fp,1), GP(l2fp,5), GP(l2fp,3), GP(l2fp,4),
          HS, ASB, ADB, AEB, ABUF, EXB, AMAX, DEN, WP, UD, UE, OUTP2);
  run_gat(stream, OUTP, NPACKET, 64, OUTP, NPACKET, 64, ei_pp, E_PP, ea_pp,
          GP(l2pp,2), GP(l2pp,0), GP(l2pp,1), GP(l2pp,5), GP(l2pp,3), GP(l2pp,4),
          HS, ASB, ADB, AEB, ABUF, EXB, AMAX, DEN, WP, UD, UE, OUTP2);
  run_gat(stream, OUTP, NPACKET, 64, OUTF, NFLOW, 64, ei_pf, E_PF, ea_pf,
          GP(l2pf,2), GP(l2pf,0), GP(l2pf,1), GP(l2pf,5), GP(l2pf,3), GP(l2pf,4),
          HS, ASB, ADB, AEB, ABUF, EXB, AMAX, DEN, WP, UD, UE, OUTF2);
  run_bn(stream, OUTP2, NPACKET, bn2_pg, bn2_pb, SUMS);
  run_bn(stream, OUTF2, NFLOW,   bn2_fg, bn2_fb, SUMS);

  // ---- global mean pool per node type, concat ----
  fillN(stream, POOLF, 0u, (long)NGRAPH * H_DIM);
  fillN(stream, CNTF,  0u, NGRAPH);
  fillN(stream, POOLP, 0u, (long)NGRAPH * H_DIM);
  fillN(stream, CNTP,  0u, NGRAPH);
  pool_scatter<<<NFLOW / 8, 256, 0, stream>>>(OUTF2, batch_flow, POOLF, CNTF, NFLOW);
  pool_scatter<<<NPACKET / 8, 256, 0, stream>>>(OUTP2, batch_packet, POOLP, CNTP, NPACKET);
  emb_build<<<(NGRAPH * 128) / 256, 256, 0, stream>>>(POOLF, CNTF, POOLP, CNTP, EMB);

  // ---- MLP (linear chain collapsed to 128x8) + log_softmax ----
  mlp_t1<<<8, 256, 0, stream>>>(mW1, mW2, T1);
  mlp_weff<<<4, 256, 0, stream>>>(T1, mW3, WEFF);
  mlp_beff<<<1, 8, 0, stream>>>(mb1, mW2, mb2, mW3, mb3, BEFF);
  mlp_final<<<NGRAPH / 256, 256, 0, stream>>>(EMB, WEFF, BEFF, (float*)d_out);
#undef GP
}
